// GraphormerPooling_11819749998825
// MI455X (gfx1250) — matrix operand order repaired and checked by reading, not compiled, not run on hardware
//
#include <hip/hip_runtime.h>
#include <hip/hip_bf16.h>

#define B_   32
#define N_   256
#define D_   512
#define H_   8
#define DK_  64
#define L_   4
#define DFF_ 2048
#define S_   257
#define R_   (B_ * S_)   // 8224 rows, divisible by 16

typedef _Float16 h8   __attribute__((ext_vector_type(8)));
typedef _Float16 v16h __attribute__((ext_vector_type(16)));
typedef float    v8f  __attribute__((ext_vector_type(8)));

// ---------------- block reductions (blockDim.x == 256) ----------------
__device__ __forceinline__ float block_sum(float v, float* red) {
    int tid = threadIdx.x;
    red[tid] = v; __syncthreads();
    for (int off = 128; off > 0; off >>= 1) {
        if (tid < off) red[tid] += red[tid + off];
        __syncthreads();
    }
    float r = red[0]; __syncthreads();
    return r;
}
__device__ __forceinline__ float block_max(float v, float* red) {
    int tid = threadIdx.x;
    red[tid] = v; __syncthreads();
    for (int off = 128; off > 0; off >>= 1) {
        if (tid < off) red[tid] = fmaxf(red[tid], red[tid + off]);
        __syncthreads();
    }
    float r = red[0]; __syncthreads();
    return r;
}

// ---------------- LN over x0 = concat(cls, x); emit h in f16 ----------------
__global__ __launch_bounds__(256)
void ln1_kernel(const float* __restrict__ x, const float* __restrict__ cls,
                const float* __restrict__ g, const float* __restrict__ b,
                _Float16* __restrict__ hF16,
                float* __restrict__ hRow0F32, _Float16* __restrict__ hRow0F16) {
    __shared__ float red[256];
    int r  = blockIdx.x;          // 0..R_-1
    int bb = r / S_, s = r % S_;
    int tid = threadIdx.x;
    int d0 = tid, d1 = tid + 256;
    float v0 = (s == 0) ? cls[d0] : x[((size_t)bb * N_ + (s - 1)) * D_ + d0];
    float v1 = (s == 0) ? cls[d1] : x[((size_t)bb * N_ + (s - 1)) * D_ + d1];
    float mean = block_sum(v0 + v1, red) * (1.0f / (float)D_);
    float a0 = v0 - mean, a1 = v1 - mean;
    float var = block_sum(a0 * a0 + a1 * a1, red) * (1.0f / (float)D_);
    float rstd = rsqrtf(var + 1e-5f);
    float o0 = a0 * rstd * g[d0] + b[d0];
    float o1 = a1 * rstd * g[d1] + b[d1];
    hF16[(size_t)r * D_ + d0] = (_Float16)o0;
    hF16[(size_t)r * D_ + d1] = (_Float16)o1;
    if (s == 0) {
        hRow0F32[(size_t)bb * D_ + d0] = o0;
        hRow0F32[(size_t)bb * D_ + d1] = o1;
        hRow0F16[(size_t)bb * D_ + d0] = (_Float16)o0;
        hRow0F16[(size_t)bb * D_ + d1] = (_Float16)o1;
    }
}

// ---------------- W (KxN f32, row-major) -> WT (NxK f16, row-major) ----------------
__global__ __launch_bounds__(256)
void transpose_f16_kernel(const float* __restrict__ W, _Float16* __restrict__ WT,
                          int K, int Nn) {
    int i = blockIdx.x * 256 + threadIdx.x;
    if (i < K * Nn) {
        int k = i / Nn, n = i % Nn;
        WT[(size_t)n * K + k] = (_Float16)W[i];
    }
}

// ---------------- WMMA GEMM: C[M x N] = A(f16, lda) @ BT(f16 N x K)^T + bias ------
// one wave per 16(M) x 64(N) tile; grid must cover exactly Mt*Ncw tiles / 8 waves
__global__ __launch_bounds__(256)
void wmma_gemm_kernel(const _Float16* __restrict__ A, int lda,
                      const _Float16* __restrict__ BT,
                      const float* __restrict__ bias,
                      float* __restrict__ C, int ldc,
                      int Mt, int Ncw, int K) {
    int w    = threadIdx.x >> 5;
    int lane = threadIdx.x & 31;
    int idx  = blockIdx.x * 8 + w;
    if (idx >= Mt * Ncw) return;             // wave-uniform: EXEC stays all-ones
    int mt = idx / Ncw, nc = idx % Ncw;
    int half = lane >> 4, l16 = lane & 15;

    const _Float16* aRow = A + (size_t)(mt * 16 + l16) * lda;
    // B operand: lane holds column (nc*64 + t*16 + l16), 16 contiguous K values
    const _Float16* bp[4];
#pragma unroll
    for (int t = 0; t < 4; ++t)
        bp[t] = BT + (size_t)(nc * 64 + t * 16 + l16) * K + half * 16;

    v8f acc[4] = {v8f{}, v8f{}, v8f{}, v8f{}};
    for (int k0 = 0; k0 < K; k0 += 32) {
        __builtin_prefetch(aRow + k0 + 128, 0, 1);   // global_prefetch_b8
        union { v16h v; h8 h[2]; } au;
        au.h[0] = *(const h8*)(aRow + k0 + half * 16);       // K in [k0+half*16, +8)
        au.h[1] = *(const h8*)(aRow + k0 + half * 16 + 8);   // K in [k0+half*16+8, +8)
#pragma unroll
        for (int t = 0; t < 4; ++t) {
            v16h bv = *(const v16h*)(bp[t] + k0);
            acc[t] = __builtin_amdgcn_wmma_f32_16x16x32_f16(
                false, au.v, false, bv, (short)0, acc[t], false, false);
        }
    }
    // C layout: VGPR i -> m = i + 8*half, n = l16
    int mBase = mt * 16 + 8 * half;
#pragma unroll
    for (int t = 0; t < 4; ++t) {
        int n = nc * 64 + t * 16 + l16;
        float bv = bias ? bias[n] : 0.0f;
#pragma unroll
        for (int i = 0; i < 8; ++i)
            C[(size_t)(mBase + i) * ldc + n] = acc[t][i] + bv;
    }
}

// ---------------- attention, query row 0 only; one block per (b,h) ----------------
__global__ __launch_bounds__(256)
void attn_row0_kernel(const float* __restrict__ Q,   // B x D (row-0 q per batch)
                      const float* __restrict__ Kp,  // R x D
                      const float* __restrict__ Vp,  // R x D
                      const unsigned char* __restrict__ cmask, // B x N bool
                      const float* __restrict__ abias,         // H x S x S
                      _Float16* __restrict__ oF16) {           // B x D
    __shared__ float sc[S_];
    __shared__ float red[256];
    __shared__ float qs[DK_];
    int bh = blockIdx.x, bb = bh >> 3, hh = bh & 7;
    int tid = threadIdx.x;
    if (tid < DK_) qs[tid] = Q[(size_t)bb * D_ + hh * DK_ + tid];
    __syncthreads();
    // scores (masked): key 0 (CLS) always blocked; others follow channel_mask
    for (int s = tid; s < S_; s += 256) {
        bool keyok = (s > 0) && (cmask[bb * N_ + (s - 1)] != 0);
        float a = 0.0f;
        const float* kr = Kp + ((size_t)bb * S_ + s) * D_ + hh * DK_;
#pragma unroll 8
        for (int j = 0; j < DK_; ++j) a += qs[j] * kr[j];
        sc[s] = keyok ? a * 0.125f : -__builtin_inff();
    }
    __syncthreads();
    float lm = -__builtin_inff();
    for (int s = tid; s < S_; s += 256) lm = fmaxf(lm, sc[s]);
    float mx = block_max(lm, red);
    float ls = 0.0f;
    for (int s = tid; s < S_; s += 256) { float e = expf(sc[s] - mx); sc[s] = e; ls += e; }
    __syncthreads();
    float inv = 1.0f / block_sum(ls, red);
    for (int s = tid; s < S_; s += 256)
        sc[s] = sc[s] * inv * abias[(size_t)hh * S_ * S_ + s];   // row q=0 of bias
    __syncthreads();
    // o[d] = sum_s p[s] * V[b,s,h*64+d] ; 4 s-chunks x 64 dims
    int d = tid & 63, ch = tid >> 6;
    float a = 0.0f;
    for (int s = ch; s < S_; s += 4)
        a += sc[s] * Vp[((size_t)bb * S_ + s) * D_ + hh * DK_ + d];
    red[tid] = a; __syncthreads();
    if (ch == 0) {
        float o = red[d] + red[64 + d] + red[128 + d] + red[192 + d];
        oF16[(size_t)bb * D_ + hh * DK_ + d] = (_Float16)o;
    }
}

// ---------------- h1 = hRow0 + (o@Wo+bo); h2 = LN2(h1) as f16 ----------------
__global__ __launch_bounds__(256)
void h1_ln2_kernel(const float* __restrict__ hRow0, const float* __restrict__ woOut,
                   const float* __restrict__ g, const float* __restrict__ b,
                   float* __restrict__ h1F32, _Float16* __restrict__ h2F16) {
    __shared__ float red[256];
    int bb = blockIdx.x, tid = threadIdx.x;
    int d0 = tid, d1 = tid + 256;
    float v0 = hRow0[(size_t)bb * D_ + d0] + woOut[(size_t)bb * D_ + d0];
    float v1 = hRow0[(size_t)bb * D_ + d1] + woOut[(size_t)bb * D_ + d1];
    h1F32[(size_t)bb * D_ + d0] = v0;
    h1F32[(size_t)bb * D_ + d1] = v1;
    float mean = block_sum(v0 + v1, red) * (1.0f / (float)D_);
    float a0 = v0 - mean, a1 = v1 - mean;
    float var = block_sum(a0 * a0 + a1 * a1, red) * (1.0f / (float)D_);
    float rstd = rsqrtf(var + 1e-5f);
    h2F16[(size_t)bb * D_ + d0] = (_Float16)(a0 * rstd * g[d0] + b[d0]);
    h2F16[(size_t)bb * D_ + d1] = (_Float16)(a1 * rstd * g[d1] + b[d1]);
}

// ---------------- exact GELU -> f16 ----------------
__global__ __launch_bounds__(256)
void gelu_kernel(const float* __restrict__ gin, _Float16* __restrict__ gout, int n) {
    int i = blockIdx.x * 256 + threadIdx.x;
    if (i < n) {
        float v = gin[i];
        gout[i] = (_Float16)(0.5f * v * (1.0f + erff(v * 0.70710678118654752f)));
    }
}

// ---------------- out = LN_f(h1 + ff) ----------------
__global__ __launch_bounds__(256)
void final_kernel(const float* __restrict__ h1F32, const float* __restrict__ ffOut,
                  const float* __restrict__ g, const float* __restrict__ b,
                  float* __restrict__ out) {
    __shared__ float red[256];
    int bb = blockIdx.x, tid = threadIdx.x;
    int d0 = tid, d1 = tid + 256;
    float v0 = h1F32[(size_t)bb * D_ + d0] + ffOut[(size_t)bb * D_ + d0];
    float v1 = h1F32[(size_t)bb * D_ + d1] + ffOut[(size_t)bb * D_ + d1];
    float mean = block_sum(v0 + v1, red) * (1.0f / (float)D_);
    float a0 = v0 - mean, a1 = v1 - mean;
    float var = block_sum(a0 * a0 + a1 * a1, red) * (1.0f / (float)D_);
    float rstd = rsqrtf(var + 1e-5f);
    out[(size_t)bb * D_ + d0] = a0 * rstd * g[d0] + b[d0];
    out[(size_t)bb * D_ + d1] = a1 * rstd * g[d1] + b[d1];
}

extern "C" void kernel_launch(void* const* d_in, const int* in_sizes, int n_in,
                              void* d_out, int out_size, void* d_ws, size_t ws_size,
                              hipStream_t stream) {
    (void)in_sizes; (void)n_in; (void)out_size; (void)ws_size;
    const float*         x     = (const float*)d_in[0];
    const unsigned char* cmask = (const unsigned char*)d_in[1];
    const float*         cls   = (const float*)d_in[2];
    const float*         abias = (const float*)d_in[3];
    const float* Wq = (const float*)d_in[4];   const float* bq = (const float*)d_in[5];
    const float* Wk = (const float*)d_in[6];   const float* bk = (const float*)d_in[7];
    const float* Wv = (const float*)d_in[8];   const float* bv = (const float*)d_in[9];
    const float* Wo = (const float*)d_in[10];  const float* bo = (const float*)d_in[11];
    const float* ln1_g = (const float*)d_in[12]; const float* ln1_b = (const float*)d_in[13];
    const float* Wf1 = (const float*)d_in[14]; const float* bf1 = (const float*)d_in[15];
    const float* Wf2 = (const float*)d_in[16]; const float* bf2 = (const float*)d_in[17];
    const float* ln2_g = (const float*)d_in[18]; const float* ln2_b = (const float*)d_in[19];
    const float* lnf_g = (const float*)d_in[20]; const float* lnf_b = (const float*)d_in[21];

    // only layer L-1 = 3 contributes to the output (reference feeds x0 to every layer)
    const int l = L_ - 1;
    const float* Wq3 = Wq + (size_t)l * D_ * D_;   const float* bq3 = bq + (size_t)l * D_;
    const float* Wk3 = Wk + (size_t)l * D_ * D_;   const float* bk3 = bk + (size_t)l * D_;
    const float* Wv3 = Wv + (size_t)l * D_ * D_;   const float* bv3 = bv + (size_t)l * D_;
    const float* Wo3 = Wo + (size_t)l * D_ * D_;   const float* bo3 = bo + (size_t)l * D_;
    const float* ln1g3 = ln1_g + (size_t)l * D_;   const float* ln1b3 = ln1_b + (size_t)l * D_;
    const float* ln2g3 = ln2_g + (size_t)l * D_;   const float* ln2b3 = ln2_b + (size_t)l * D_;
    const float* Wf1_3 = Wf1 + (size_t)l * D_ * DFF_; const float* bf1_3 = bf1 + (size_t)l * DFF_;
    const float* Wf2_3 = Wf2 + (size_t)l * DFF_ * D_; const float* bf2_3 = bf2 + (size_t)l * D_;

    // workspace carve-out (256B aligned)
    size_t off = 0;
    auto carve = [&](size_t bytes) {
        void* p = (char*)d_ws + off;
        off += (bytes + 255) & ~(size_t)255;
        return p;
    };
    _Float16* hF16     = (_Float16*)carve((size_t)R_ * D_ * 2);
    float*    hRow0F32 = (float*)   carve((size_t)B_ * D_ * 4);
    _Float16* hRow0F16 = (_Float16*)carve((size_t)B_ * D_ * 2);
    _Float16* WqT  = (_Float16*)carve((size_t)D_ * D_ * 2);
    _Float16* WkT  = (_Float16*)carve((size_t)D_ * D_ * 2);
    _Float16* WvT  = (_Float16*)carve((size_t)D_ * D_ * 2);
    _Float16* WoT  = (_Float16*)carve((size_t)D_ * D_ * 2);
    _Float16* Wf1T = (_Float16*)carve((size_t)D_ * DFF_ * 2);
    _Float16* Wf2T = (_Float16*)carve((size_t)DFF_ * D_ * 2);
    float*    Kproj = (float*)carve((size_t)R_ * D_ * 4);
    float*    Vproj = (float*)carve((size_t)R_ * D_ * 4);
    float*    Qrow0 = (float*)carve((size_t)B_ * D_ * 4);
    _Float16* oF16  = (_Float16*)carve((size_t)B_ * D_ * 2);
    float*    woOut = (float*)carve((size_t)B_ * D_ * 4);
    float*    h1F32 = (float*)carve((size_t)B_ * D_ * 4);
    _Float16* h2F16 = (_Float16*)carve((size_t)B_ * D_ * 2);
    float*    gOut  = (float*)carve((size_t)B_ * DFF_ * 4);
    _Float16* gF16  = (_Float16*)carve((size_t)B_ * DFF_ * 2);
    float*    ffOut = (float*)carve((size_t)B_ * D_ * 4);

    // 1) h = LN1(x0) for all rows (f16), plus compact CLS rows
    ln1_kernel<<<R_, 256, 0, stream>>>(x, cls, ln1g3, ln1b3, hF16, hRow0F32, hRow0F16);

    // 2) weight transposes -> f16 (B operand wants N-major, contiguous K)
    transpose_f16_kernel<<<(D_ * D_) / 256, 256, 0, stream>>>(Wq3, WqT, D_, D_);
    transpose_f16_kernel<<<(D_ * D_) / 256, 256, 0, stream>>>(Wk3, WkT, D_, D_);
    transpose_f16_kernel<<<(D_ * D_) / 256, 256, 0, stream>>>(Wv3, WvT, D_, D_);
    transpose_f16_kernel<<<(D_ * D_) / 256, 256, 0, stream>>>(Wo3, WoT, D_, D_);
    transpose_f16_kernel<<<(D_ * DFF_) / 256, 256, 0, stream>>>(Wf1_3, Wf1T, D_, DFF_);
    transpose_f16_kernel<<<(DFF_ * D_) / 256, 256, 0, stream>>>(Wf2_3, Wf2T, DFF_, D_);

    // 3) big projections: K,V over all 8224 rows; Q over 32 CLS rows
    wmma_gemm_kernel<<<(R_ / 16) * 8 / 8, 256, 0, stream>>>(hF16, D_, WkT, bk3, Kproj, D_,
                                                            R_ / 16, D_ / 64, D_);
    wmma_gemm_kernel<<<(R_ / 16) * 8 / 8, 256, 0, stream>>>(hF16, D_, WvT, bv3, Vproj, D_,
                                                            R_ / 16, D_ / 64, D_);
    wmma_gemm_kernel<<<2 * 8 / 8, 256, 0, stream>>>(hRow0F16, D_, WqT, bq3, Qrow0, D_,
                                                    B_ / 16, D_ / 64, D_);

    // 4) attention for query row 0 (one block per (b,h))
    attn_row0_kernel<<<B_ * H_, 256, 0, stream>>>(Qrow0, Kproj, Vproj, cmask, abias, oF16);

    // 5) o @ Wo + bo ; h1 = h + .. ; h2 = LN2(h1)
    wmma_gemm_kernel<<<2 * 8 / 8, 256, 0, stream>>>(oF16, D_, WoT, bo3, woOut, D_,
                                                    B_ / 16, D_ / 64, D_);
    h1_ln2_kernel<<<B_, 256, 0, stream>>>(hRow0F32, woOut, ln2g3, ln2b3, h1F32, h2F16);

    // 6) FFN: gelu(h2 @ Wf1 + bf1) @ Wf2 + bf2
    wmma_gemm_kernel<<<2 * 32 / 8, 256, 0, stream>>>(h2F16, D_, Wf1T, bf1_3, gOut, DFF_,
                                                     B_ / 16, DFF_ / 64, D_);
    gelu_kernel<<<(B_ * DFF_) / 256, 256, 0, stream>>>(gOut, gF16, B_ * DFF_);
    wmma_gemm_kernel<<<2 * 8 / 8, 256, 0, stream>>>(gF16, DFF_, Wf2T, bf2_3, ffOut, D_,
                                                    B_ / 16, D_ / 64, DFF_);

    // 7) out = LN_f(h1 + ff) -> (B, D) f32
    final_kernel<<<B_, 256, 0, stream>>>(h1F32, ffOut, lnf_g, lnf_b, (float*)d_out);
}
